// UpPolyAct_77644418777723
// MI455X (gfx1250) — matrix operand order repaired and checked by reading, not compiled, not run on hardware
//
#include <hip/hip_runtime.h>

// ---------------------------------------------------------------------------
// UpsampleRFFT(2x) -> PolyAct(deg2) -> LPF_RFFT -> decimate(2x) as four
// bf16 WMMA GEMMs per 64x64 image, transpose-parity chosen so every WMMA
// fragment access is a contiguous 16B/32B vector op:
//
//   R1 = x^T A^T   (=t1^T)  -> natural C write == t1  row-major   (LDS)
//   u  = t1  A^T   -> poly  -> natural C write == p^T row-major   (LDS)
//   R3 = p^T D^T   (=t2^T)  -> natural C write == t2  row-major   (LDS)
//   R4 = D t2^T    (=out^T) -> natural C write == out row-major   (global f32)
//
// Per-wave the B operand column block is fixed, so B fragments are loaded
// once per phase and the per-wave tile loops are fully unrolled to give the
// scheduler independent WMMA chains (fills the 4-coexec bf16 WMMA hazard).
// ~26 GFLOP vs 128 MB HBM (~200 FLOP/B) => matrix pipe + wide loads.
// ---------------------------------------------------------------------------

typedef __attribute__((ext_vector_type(16))) __bf16 v16bf;
typedef __attribute__((ext_vector_type(8)))  __bf16 v8bf;
typedef __attribute__((ext_vector_type(8)))  float  v8f;

struct Q2 { uint4 a, b; };          // 32 bytes == one v16bf fragment

#define PI_F 3.14159265358979323846f

// A[n][m] = (1/64) * (1 + 2*sum_{k=1..32} cos(pi*k*(n-2m)/64))
__device__ inline float recon_entry(int n, int m) {
    const int d = n - 2 * m;
    float s = 1.0f;
#pragma unroll
    for (int k = 1; k <= 32; ++k)
        s += 2.0f * __cosf(PI_F * (float)(k * d) * (1.0f / 64.0f));
    return s * (1.0f / 64.0f);
}

// D[m][n] = (1/128) * (1 + 2*sum_{k=1..31} cos(pi*k*(2m-n)/64) + cos(pi*(2m-n)/2))
__device__ inline float lpfdec_entry(int m, int n) {
    const int d = 2 * m - n;
    float s = 1.0f;
#pragma unroll
    for (int k = 1; k <= 31; ++k)
        s += 2.0f * __cosf(PI_F * (float)(k * d) * (1.0f / 64.0f));
    s += __cosf(PI_F * (float)d * 0.5f);
    return s * (1.0f / 128.0f);
}

// ws layout (bf16): [0,8192) = A (128x64 row-major), [8192,16384) = D (64x128)
__global__ __launch_bounds__(256) void build_filter_mats(__bf16* __restrict__ ws) {
    const int gid = blockIdx.x * blockDim.x + threadIdx.x;   // 0 .. 16383
    if (gid < 8192) {
        const int n = gid >> 6, m = gid & 63;
        ws[gid] = (__bf16)recon_entry(n, m);
    } else {
        const int j = gid - 8192;
        const int m = j >> 7, n = j & 127;
        ws[gid] = (__bf16)lpfdec_entry(m, n);
    }
}

// A fragment (16x32 tile, row-major source): two contiguous 16B runs per lane.
__device__ inline v16bf load_a_frag(const __bf16* base, int ld, int row0, int k0, int lane) {
    const int m  = row0 + (lane & 15);
    const int hi = (lane >> 4) & 1;
    const uint4* p = reinterpret_cast<const uint4*>(base + m * ld + k0);
    Q2 q;
    q.a = p[hi];        // halves K[8*hi .. 8*hi+7]
    q.b = p[2 + hi];    // halves K[16+8*hi .. 23+8*hi]
    return __builtin_bit_cast(v16bf, q);
}

// B fragment from B^T stored row-major: two contiguous 16B runs per lane.
__device__ inline v16bf load_bt_frag(const __bf16* bt, int ld, int k0, int col0, int lane) {
    const int n  = col0 + (lane & 15);
    const int hi = (lane >> 4) & 1;
    const uint4* p = reinterpret_cast<const uint4*>(bt + n * ld + k0);
    Q2 q;
    q.a = p[hi];
    q.b = p[2 + hi];
    return __builtin_bit_cast(v16bf, q);
}

// Natural C-tile store: writes result^T row-major, 8 contiguous bf16 per lane.
__device__ inline void store_ct_bf16(__bf16* dstT, int ld, int row0, int col0,
                                     int lane, const v8f& acc) {
    const int col = col0 + (lane & 15);
    const int hi  = (lane >> 4) & 1;
    v8bf h;
#pragma unroll
    for (int r = 0; r < 8; ++r) h[r] = (__bf16)acc[r];
    *reinterpret_cast<uint4*>(dstT + col * ld + row0 + 8 * hi) =
        __builtin_bit_cast(uint4, h);
}

__global__ __launch_bounds__(256) void up_poly_down(const float* __restrict__ x,
                                                    const float* __restrict__ coef,
                                                    const __bf16* __restrict__ ws,
                                                    float* __restrict__ out) {
    __shared__ __bf16 sXT[64 * 64];        // x^T  [w][h]              8 KB
    __shared__ __bf16 sScr[128 * 64];      // t1 [h'][w] / t2 [h][w'] 16 KB
    __shared__ __bf16 sPT[128 * 128];      // p^T [w'][h']            32 KB

    const __bf16* gA = ws;                 // 128x64 [h'][h]
    const __bf16* gD = ws + 8192;          // 64x128 [h][h']

    const int tid  = threadIdx.x;
    const int wave = tid >> 5;             // 0..7
    const int lane = tid & 31;
    const int hi   = (lane >> 4) & 1;
    const int nlo  = lane & 15;

    __builtin_prefetch(gA, 0, 3);
    __builtin_prefetch(gD, 0, 3);

    const float c0 = coef[0], c1 = coef[1], c2 = coef[2];

    const float* xim = x   + (size_t)blockIdx.x * 4096;
    float*       oim = out + (size_t)blockIdx.x * 4096;

    // stage x transposed into LDS (f32 -> bf16): coalesced b128 reads.
#pragma unroll
    for (int i0 = 0; i0 < 4; ++i0) {
        const int i = i0 * 256 + tid;
        const float4 v = reinterpret_cast<const float4*>(xim)[i];
        const int h = i >> 4, w0 = (i & 15) << 2;
        sXT[(w0 + 0) * 64 + h] = (__bf16)v.x;
        sXT[(w0 + 1) * 64 + h] = (__bf16)v.y;
        sXT[(w0 + 2) * 64 + h] = (__bf16)v.z;
        sXT[(w0 + 3) * 64 + h] = (__bf16)v.w;
    }

    // B fragments from gA for P1 and P2 (identical in both): col block = wave
    const v16bf bA0 = load_bt_frag(gA, 64, 0,  wave * 16, lane);
    const v16bf bA1 = load_bt_frag(gA, 64, 32, wave * 16, lane);
    __syncthreads();

    // ---- P1: R1 = x^T(64x64) @ A^T -> natural write = t1 row-major --------
    // M=64 (w), N=128 (h'), K=64 (h); wave owns ni=wave, mi=0..3.
#pragma unroll
    for (int mi = 0; mi < 4; ++mi) {
        v8f acc = {};
        v16bf a0 = load_a_frag(sXT, 64, mi * 16, 0,  lane);
        v16bf a1 = load_a_frag(sXT, 64, mi * 16, 32, lane);
        acc = __builtin_amdgcn_wmma_f32_16x16x32_bf16(false, a0, false, bA0,
                                                      (short)0, acc, false, false);
        acc = __builtin_amdgcn_wmma_f32_16x16x32_bf16(false, a1, false, bA1,
                                                      (short)0, acc, false, false);
        store_ct_bf16(sScr, 64, mi * 16, wave * 16, lane, acc);   // t1 [h'][w]
    }
    __syncthreads();

    // ---- P2: u = t1(128x64) @ A^T; poly; natural write = p^T row-major ----
    // M=128 (h'), N=128 (w'), K=64 (w); wave owns ni=wave, mi=0..7.
#pragma unroll
    for (int mi = 0; mi < 8; ++mi) {
        v8f acc = {};
        v16bf a0 = load_a_frag(sScr, 64, mi * 16, 0,  lane);
        v16bf a1 = load_a_frag(sScr, 64, mi * 16, 32, lane);
        acc = __builtin_amdgcn_wmma_f32_16x16x32_bf16(false, a0, false, bA0,
                                                      (short)0, acc, false, false);
        acc = __builtin_amdgcn_wmma_f32_16x16x32_bf16(false, a1, false, bA1,
                                                      (short)0, acc, false, false);
        v8f p;
#pragma unroll
        for (int r = 0; r < 8; ++r) {
            const float u = acc[r];
            p[r] = c0 + c1 * u + c2 * u * u;
        }
        store_ct_bf16(sPT, 128, mi * 16, wave * 16, lane, p);     // p^T [w'][h']
    }
    __syncthreads();

    // ---- P3: R3 = p^T(128x128) @ D^T -> natural write = t2 row-major ------
    // M=128 (w'), N=64 (h), K=128 (h'); wave owns ni=wave&3, mi in 4-block.
    {
        const int ni = (wave & 3) * 16;
        const int mbase = (wave >> 2) * 4;
        v16bf bD0 = load_bt_frag(gD, 128, 0,  ni, lane);
        v16bf bD1 = load_bt_frag(gD, 128, 32, ni, lane);
        v16bf bD2 = load_bt_frag(gD, 128, 64, ni, lane);
        v16bf bD3 = load_bt_frag(gD, 128, 96, ni, lane);
#pragma unroll
        for (int i = 0; i < 4; ++i) {
            const int mi = mbase + i;
            v8f acc = {};
            v16bf a0 = load_a_frag(sPT, 128, mi * 16, 0,  lane);
            v16bf a1 = load_a_frag(sPT, 128, mi * 16, 32, lane);
            v16bf a2 = load_a_frag(sPT, 128, mi * 16, 64, lane);
            v16bf a3 = load_a_frag(sPT, 128, mi * 16, 96, lane);
            acc = __builtin_amdgcn_wmma_f32_16x16x32_bf16(false, a0, false, bD0,
                                                          (short)0, acc, false, false);
            acc = __builtin_amdgcn_wmma_f32_16x16x32_bf16(false, a1, false, bD1,
                                                          (short)0, acc, false, false);
            acc = __builtin_amdgcn_wmma_f32_16x16x32_bf16(false, a2, false, bD2,
                                                          (short)0, acc, false, false);
            acc = __builtin_amdgcn_wmma_f32_16x16x32_bf16(false, a3, false, bD3,
                                                          (short)0, acc, false, false);
            store_ct_bf16(sScr, 128, mi * 16, ni, lane, acc);     // t2 [h][w']
        }
    }
    __syncthreads();

    // ---- P4: R4 = D(64x128) @ t2^T = out^T -> natural write = out ---------
    // M=64 (w), N=64 (h), K=128 (w'); wave owns ni=wave&3, mi in 2-block.
    {
        const int ni = (wave & 3) * 16;
        const int mbase = (wave >> 2) * 2;
        v16bf b0 = load_bt_frag(sScr, 128, 0,  ni, lane);
        v16bf b1 = load_bt_frag(sScr, 128, 32, ni, lane);
        v16bf b2 = load_bt_frag(sScr, 128, 64, ni, lane);
        v16bf b3 = load_bt_frag(sScr, 128, 96, ni, lane);
#pragma unroll
        for (int i = 0; i < 2; ++i) {
            const int mi = mbase + i;
            v8f acc = {};
            v16bf a0 = load_a_frag(gD, 128, mi * 16, 0,  lane);
            v16bf a1 = load_a_frag(gD, 128, mi * 16, 32, lane);
            v16bf a2 = load_a_frag(gD, 128, mi * 16, 64, lane);
            v16bf a3 = load_a_frag(gD, 128, mi * 16, 96, lane);
            acc = __builtin_amdgcn_wmma_f32_16x16x32_bf16(false, a0, false, b0,
                                                          (short)0, acc, false, false);
            acc = __builtin_amdgcn_wmma_f32_16x16x32_bf16(false, a1, false, b1,
                                                          (short)0, acc, false, false);
            acc = __builtin_amdgcn_wmma_f32_16x16x32_bf16(false, a2, false, b2,
                                                          (short)0, acc, false, false);
            acc = __builtin_amdgcn_wmma_f32_16x16x32_bf16(false, a3, false, b3,
                                                          (short)0, acc, false, false);
            // natural write of (out^T)^T == out row-major: 8 contiguous f32/lane
            const int col = ni + nlo;                // h
            const int row = mi * 16 + 8 * hi;        // w
            float4* o4 = reinterpret_cast<float4*>(oim + col * 64 + row);
            o4[0] = make_float4(acc[0], acc[1], acc[2], acc[3]);
            o4[1] = make_float4(acc[4], acc[5], acc[6], acc[7]);
        }
    }
}

extern "C" void kernel_launch(void* const* d_in, const int* in_sizes, int n_in,
                              void* d_out, int out_size, void* d_ws, size_t ws_size,
                              hipStream_t stream) {
    const float* x    = (const float*)d_in[0];
    const float* coef = (const float*)d_in[1];
    float*       out  = (float*)d_out;
    __bf16*      wsb  = (__bf16*)d_ws;   // needs 32 KB

    build_filter_mats<<<64, 256, 0, stream>>>(wsb);

    const int n_images = in_sizes[0] / 4096;   // B*C = 4096 images of 64x64
    up_poly_down<<<n_images, 256, 0, stream>>>(x, coef, wsb, out);
}
